// MultiheadLeap_59648505807027
// MI455X (gfx1250) — matrix operand order repaired and checked by reading, not compiled, not run on hardware
//
#include <hip/hip_runtime.h>
#include <hip/hip_bf16.h>
#include <math.h>

typedef __attribute__((ext_vector_type(16))) _Float16 v16h;
typedef __attribute__((ext_vector_type(8)))  float    v8f;

constexpr int B_      = 4;
constexpr int S_      = 4096;
constexpr int HID_    = 1024;
constexpr int NH_     = 16;
constexpr int HEAD_   = 64;
constexpr int WIN_    = 512;
constexpr float SCALE_ = 0.125f;   // 1/sqrt(64)
constexpr float EPS_   = 1e-5f;

// ---------------------------------------------------------------------------
// Kernel A: focus weights  w[b,h,s] = exp(SCALE * dot(f[b,s,h,:], k[b,s,h,:]))
// ---------------------------------------------------------------------------
__global__ __launch_bounds__(256)
void k_weights(const float* __restrict__ f, const float* __restrict__ k,
               float* __restrict__ w) {
    int idx = blockIdx.x * 256 + threadIdx.x;       // B*S*NH = 262144
    int h = idx & (NH_ - 1);
    int s = (idx >> 4) & (S_ - 1);
    int b = idx >> 16;                              // S*NH = 65536
    size_t base = ((size_t)(b * S_ + s)) * HID_ + h * HEAD_;
    const float4* fp = (const float4*)(f + base);
    const float4* kp = (const float4*)(k + base);
    float dot = 0.f;
#pragma unroll
    for (int j = 0; j < 16; ++j) {
        float4 a = fp[j], c = kp[j];
        dot += a.x * c.x + a.y * c.y + a.z * c.z + a.w * c.w;
    }
    w[((size_t)(b * NH_ + h)) * S_ + s] = __expf(dot * SCALE_);
}

// ---------------------------------------------------------------------------
// Kernel B: wv[b,h,d,s] = (f16)(w[b,h,s] * v[b,s,h,d]); LDS transpose so both
// the v reads and the wv writes are coalesced. 64(s) x 64(d) tile per block.
// ---------------------------------------------------------------------------
__global__ __launch_bounds__(256)
void k_wv(const float* __restrict__ v, const float* __restrict__ w,
          _Float16* __restrict__ wv) {
    __shared__ _Float16 lds[64][72];
    const int bh = blockIdx.y;                      // b*NH + h
    const int b  = bh >> 4, h = bh & 15;
    const int s0 = blockIdx.x * 64;
    const int t  = threadIdx.x;
    const float* wp = w + (size_t)bh * S_ + s0;
#pragma unroll
    for (int j = 0; j < 16; ++j) {
        int idx = t + 256 * j;                      // d fastest -> coalesced v
        int d = idx & 63, sl = idx >> 6;
        float val = wp[sl] * v[((size_t)(b * S_ + s0 + sl)) * HID_ + h * HEAD_ + d];
        lds[d][sl] = (_Float16)val;
    }
    __syncthreads();
#pragma unroll
    for (int j = 0; j < 16; ++j) {
        int idx = t + 256 * j;                      // s fastest -> coalesced wv
        int sl = idx & 63, d = idx >> 6;
        wv[((size_t)bh * HEAD_ + d) * S_ + s0 + sl] = lds[d][sl];
    }
}

// ---------------------------------------------------------------------------
// Kernel C: windowed denominator via block prefix scan.
// inv_den[b,h,s] = 1 / (cumsum(w)[s] - cumsum(w)[s-512] + eps)
// ---------------------------------------------------------------------------
__global__ __launch_bounds__(256)
void k_den(const float* __restrict__ w, float* __restrict__ inv_den) {
    __shared__ float buf[S_];
    __shared__ float aux[256];
    const int bh = blockIdx.x;
    const int t  = threadIdx.x;
    const float* wp = w + (size_t)bh * S_;
#pragma unroll
    for (int j = 0; j < 16; ++j) buf[t + 256 * j] = wp[t + 256 * j];
    __syncthreads();
    float loc[16];
    float run = 0.f;
#pragma unroll
    for (int j = 0; j < 16; ++j) { run += buf[t * 16 + j]; loc[j] = run; }
    aux[t] = run;
    __syncthreads();
    for (int off = 1; off < 256; off <<= 1) {
        float add = (t >= off) ? aux[t - off] : 0.f;
        __syncthreads();
        aux[t] += add;
        __syncthreads();
    }
    float base = (t > 0) ? aux[t - 1] : 0.f;
#pragma unroll
    for (int j = 0; j < 16; ++j) buf[t * 16 + j] = loc[j] + base;
    __syncthreads();
    float* ip = inv_den + (size_t)bh * S_;
#pragma unroll
    for (int j = 0; j < 16; ++j) {
        int s = t + 256 * j;
        float P  = buf[s];
        float Pm = (s >= WIN_) ? buf[s - WIN_] : 0.f;
        ip[s] = 1.0f / (P - Pm + EPS_);
    }
}

// ---------------------------------------------------------------------------
// Kernel D: banded matmul  O[s,d] = inv_den[s] * sum_u A[s,u] * wv[d,u]
// One wave per 16(s) x 16(d) tile; 17 K-tiles of 32.
// Fast path (s0 >= 512): A masks hoisted out of the loop
//   t=0     -> a_lo : element kl >= m+1
//   t=1..15 -> all-ones (fully unrolled: loop body = 2x b128 load + wmma)
//   t=16    -> a_hi : element kl <= m
// Slow path (s0 < 512): generic per-element predicate incl. u >= 0 clip.
// Layouts (wave32):
//   A 16x32 f16: lane L row m=L&15; elem e -> K = (L<16?0:16)+e
//   B 32x16 f16: lane L col n=L&15, K = (L<16?0:16)+e  (16 contiguous u)
//   C/D 16x16 f32: VGPR r, lanes 0-15 -> M=r, lanes 16-31 -> M=r+8; N=lane&15
// ---------------------------------------------------------------------------
__device__ __forceinline__ int klane(int e, int hi) {
    return (hi ? 16 : 0) + e;
}

__global__ __launch_bounds__(128)
void k_wmma(const _Float16* __restrict__ wv, const float* __restrict__ inv_den,
            float* __restrict__ out) {
    const int s0   = blockIdx.x * 16;
    const int bh   = blockIdx.y;                    // b*NH + h
    const int b    = bh >> 4, h = bh & 15;
    const int dtle = threadIdx.x >> 5;              // wave id = d-tile 0..3
    const int lane = threadIdx.x & 31;
    const int hi   = lane >> 4;
    const int lm   = lane & 15;

    const _Float16* wvrow = wv + ((size_t)bh * HEAD_ + dtle * 16 + lm) * S_;
    const int uchunk = hi * 16;                     // this lane's K-chunk base

    v8f acc = {};

    if (s0 >= WIN_) {
        // ---- fast path: no u>=0 clipping anywhere, no B-load guards ----
        v16h a_ones, a_lo;
#pragma unroll
        for (int e = 0; e < 16; ++e) {
            int kl = klane(e, hi);
            a_ones[e] = (_Float16)1.0f;
            a_lo[e]   = (_Float16)((kl >= lm + 1) ? 1.0f : 0.0f);
        }
        // t = 0: lower boundary tile, u in [s0-512, s0-481]
        {
            v16h bv = *(const v16h*)(wvrow + (s0 - WIN_) + uchunk);
            acc = __builtin_amdgcn_wmma_f32_16x16x32_f16(
                false, a_lo, false, bv, (short)0, acc, false, false);
        }
        // t = 1..15: interior all-ones tiles
#pragma unroll
        for (int t = 1; t < 16; ++t) {
            const _Float16* p = wvrow + (s0 - WIN_ + t * 32) + uchunk;
            v16h bv = *(const v16h*)p;
            __builtin_prefetch(p + 256, 0, 1);      // global_prefetch ahead
            acc = __builtin_amdgcn_wmma_f32_16x16x32_f16(
                false, a_ones, false, bv, (short)0, acc, false, false);
        }
    } else {
        // ---- slow path: sequence start, generic masks with u >= 0 clip ----
        const int srow = s0 + lm;
#pragma unroll 1
        for (int t = 0; t < 16; ++t) {
            const int u0 = s0 - WIN_ + t * 32;
            if (u0 + 31 < 0) continue;
            v16h a;
#pragma unroll
            for (int e = 0; e < 16; ++e) {
                int u = u0 + klane(e, hi);
                a[e] = (_Float16)(((u >= 0) & (u >= srow - (WIN_ - 1)))
                                      ? 1.0f : 0.0f);   // u <= srow auto (t<16)
            }
            const int uc = u0 + uchunk;
            v16h bv = {};
            if (uc >= 0) bv = *(const v16h*)(wvrow + uc);
            acc = __builtin_amdgcn_wmma_f32_16x16x32_f16(
                false, a, false, bv, (short)0, acc, false, false);
        }
    }

    // t = 16: upper boundary tile, u in [s0, s0+31], mask kl <= m
    {
        v16h a_hi;
#pragma unroll
        for (int e = 0; e < 16; ++e)
            a_hi[e] = (_Float16)((klane(e, hi) <= lm) ? 1.0f : 0.0f);
        const int uc = s0 + uchunk;
        v16h bv = {};
        if (uc < S_) bv = *(const v16h*)(wvrow + uc);  // clip at sequence end
        acc = __builtin_amdgcn_wmma_f32_16x16x32_f16(
            false, a_hi, false, bv, (short)0, acc, false, false);
    }

    // normalize rows and store (C/D f32 layout)
    const float* dnv = inv_den + (size_t)bh * S_;
    float* op = out + (size_t)b * S_ * HID_ + h * HEAD_ + dtle * 16 + lm;
#pragma unroll
    for (int r = 0; r < 8; ++r) {
        int sr = s0 + r + hi * 8;
        op[(size_t)sr * HID_] = acc[r] * dnv[sr];
    }
}

// ---------------------------------------------------------------------------
extern "C" void kernel_launch(void* const* d_in, const int* in_sizes, int n_in,
                              void* d_out, int out_size, void* d_ws, size_t ws_size,
                              hipStream_t stream) {
    // setup_inputs order: q (dead code in reference), f, k, v
    const float* f = (const float*)d_in[1];
    const float* k = (const float*)d_in[2];
    const float* v = (const float*)d_in[3];
    float* out = (float*)d_out;

    char* ws = (char*)d_ws;
    _Float16* wv      = (_Float16*)ws;                                    // 32 MB
    float*    w       = (float*)(ws + (size_t)B_ * NH_ * HEAD_ * S_ * 2); // 1 MB
    float*    inv_den = w + (size_t)B_ * NH_ * S_;                        // 1 MB

    k_weights<<<(B_ * S_ * NH_) / 256, 256, 0, stream>>>(f, k, w);
    k_wv<<<dim3(S_ / 64, B_ * NH_), 256, 0, stream>>>(v, w, wv);
    k_den<<<B_ * NH_, 256, 0, stream>>>(w, inv_den);
    k_wmma<<<dim3(S_ / 16, B_ * NH_), 128, 0, stream>>>(wv, inv_den, out);
}
